// StreamingTransformerLayer_23562190586520
// MI455X (gfx1250) — matrix operand order repaired and checked by reading, not compiled, not run on hardware
//
#include <hip/hip_runtime.h>
#include <math.h>

// ---------------------------------------------------------------------------
// Problem constants (from reference setup_inputs)
// ---------------------------------------------------------------------------
#define BB   2
#define TT   1024
#define CC   2048
#define FF   8192
#define HH   32
#define DD   64
#define CAP  1024
#define DS   (DD/2)
#define EPS  1e-5f

typedef __attribute__((ext_vector_type(16))) __bf16 v16bf;
typedef __attribute__((ext_vector_type(8)))  float  v8f;

union FragU { uint4 u[2]; v16bf v; };

static __device__ __forceinline__ v8f vzero8() {
    v8f z;
#pragma unroll
    for (int i = 0; i < 8; ++i) z[i] = 0.0f;
    return z;
}

static __device__ __forceinline__ v8f wmma_bf16(v16bf a, v16bf b, v8f c) {
    return __builtin_amdgcn_wmma_f32_16x16x32_bf16(false, a, false, b,
                                                   (short)0, c, false, false);
}

static __device__ __forceinline__ __bf16 f2bf(float f) { return (__bf16)f; }

// async 16B global -> LDS copy (per-lane), tracked by ASYNCcnt
static __device__ __forceinline__ void async_g2l_b128(const void* gaddr, void* lds) {
    unsigned loff = (unsigned)(uintptr_t)lds;   // low 32 bits of flat LDS addr = LDS offset
    asm volatile("global_load_async_to_lds_b128 %0, %1, off"
                 :: "v"(loff), "v"(gaddr) : "memory");
}
static __device__ __forceinline__ void wait_asynccnt0() {
    asm volatile("s_wait_asynccnt 0x0" ::: "memory");
}

// ---------------------------------------------------------------------------
// fp32 -> bf16 elementwise convert (weights, cache)
// ---------------------------------------------------------------------------
__global__ __launch_bounds__(256) void cvt_f32_bf16(const float* __restrict__ src,
                                                    __bf16* __restrict__ dst, int n) {
    for (int i = blockIdx.x * 256 + threadIdx.x; i < n; i += gridDim.x * 256)
        dst[i] = f2bf(src[i]);
}

// ---------------------------------------------------------------------------
// LayerNorm: one block per row (C=2048, 256 threads, 8 elems/thread) -> bf16
// ---------------------------------------------------------------------------
__global__ __launch_bounds__(256) void layernorm_bf16(const float* __restrict__ x,
                                                      const float* __restrict__ w,
                                                      const float* __restrict__ b,
                                                      __bf16* __restrict__ out) {
    __shared__ float s1[256], s2[256];
    const int row = blockIdx.x;
    const float* xr = x + (size_t)row * CC;
    float sum = 0.f, sq = 0.f;
    float v[CC / 256];
#pragma unroll
    for (int i = 0; i < CC / 256; ++i) {
        float t = xr[threadIdx.x + 256 * i];
        v[i] = t; sum += t; sq += t * t;
    }
    s1[threadIdx.x] = sum; s2[threadIdx.x] = sq;
    __syncthreads();
    for (int off = 128; off > 0; off >>= 1) {
        if (threadIdx.x < off) {
            s1[threadIdx.x] += s1[threadIdx.x + off];
            s2[threadIdx.x] += s2[threadIdx.x + off];
        }
        __syncthreads();
    }
    const float mean = s1[0] * (1.0f / CC);
    const float var  = s2[0] * (1.0f / CC) - mean * mean;
    const float rs   = rsqrtf(var + EPS);
    __bf16* orow = out + (size_t)row * CC;
#pragma unroll
    for (int i = 0; i < CC / 256; ++i) {
        int c = threadIdx.x + 256 * i;
        orow[c] = f2bf((v[i] - mean) * rs * w[c] + b[c]);
    }
}

// ---------------------------------------------------------------------------
// Generic bf16 WMMA GEMM:  Out[M,N] = A[M,K] * W[N,K]^T   (+ epilogue)
// Block: 256 thr = 8 waves; tile 128x128x32; wave -> 32x64 (2x4 WMMA tiles)
// Tiles staged via GLOBAL_LOAD_ASYNC_TO_LDS_B128 (ASYNCcnt path).
// EPI: 0 = store f32, 1 = Res + acc -> f32, 2 = gelu(acc) -> bf16
// ---------------------------------------------------------------------------
template <int EPI>
__global__ __launch_bounds__(256) void gemm_bf16(const __bf16* __restrict__ A,
                                                 const __bf16* __restrict__ W,
                                                 float* __restrict__ OutF,
                                                 __bf16* __restrict__ OutB,
                                                 const float* __restrict__ Res,
                                                 int M, int N, int K) {
    __shared__ __bf16 As[128][40];
    __shared__ __bf16 Bs[128][40];

    const int bn = blockIdx.x * 128;
    const int bm = blockIdx.y * 128;
    const int tid  = threadIdx.x;
    const int wave = tid >> 5;
    const int lane = tid & 31;
    const int lr = lane & 15, lh = lane >> 4;
    const int wm = (wave & 3) * 32;   // 4 waves along M
    const int wn = (wave >> 2) * 64;  // 2 waves along N

    v8f acc[2][4];
#pragma unroll
    for (int i = 0; i < 2; ++i)
#pragma unroll
        for (int j = 0; j < 4; ++j) acc[i][j] = vzero8();

    for (int k0 = 0; k0 < K; k0 += 32) {
#pragma unroll
        for (int p = 0; p < 2; ++p) {
            int i = tid + 256 * p;
            int row = i >> 2, c4 = i & 3;
            async_g2l_b128(&A[(size_t)(bm + row) * K + k0 + c4 * 8], &As[row][c4 * 8]);
            async_g2l_b128(&W[(size_t)(bn + row) * K + k0 + c4 * 8], &Bs[row][c4 * 8]);
        }
        if (k0 + 32 < K) {  // pull next tiles toward L2 (global_prefetch_b8)
            __builtin_prefetch(&A[(size_t)(bm + (tid >> 1)) * K + k0 + 32], 0, 1);
            __builtin_prefetch(&W[(size_t)(bn + (tid >> 1)) * K + k0 + 32], 0, 1);
        }
        wait_asynccnt0();
        __syncthreads();

        FragU fa[2], fb[4];
#pragma unroll
        for (int mi = 0; mi < 2; ++mi) {
            const __bf16* ap = &As[wm + mi * 16 + lr][8 * lh];
            fa[mi].u[0] = *(const uint4*)ap;
            fa[mi].u[1] = *(const uint4*)(ap + 16);
        }
#pragma unroll
        for (int ni = 0; ni < 4; ++ni) {
            const __bf16* bp = &Bs[wn + ni * 16 + lr][8 * lh];
            fb[ni].u[0] = *(const uint4*)bp;
            fb[ni].u[1] = *(const uint4*)(bp + 16);
        }
#pragma unroll
        for (int mi = 0; mi < 2; ++mi)
#pragma unroll
            for (int ni = 0; ni < 4; ++ni)
                acc[mi][ni] = wmma_bf16(fa[mi].v, fb[ni].v, acc[mi][ni]);
        __syncthreads();
    }

#pragma unroll
    for (int mi = 0; mi < 2; ++mi)
#pragma unroll
        for (int ni = 0; ni < 4; ++ni)
#pragma unroll
            for (int r = 0; r < 8; ++r) {
                int row = bm + wm + mi * 16 + r + 8 * lh;
                int col = bn + wn + ni * 16 + lr;
                float val = acc[mi][ni][r];
                size_t idx = (size_t)row * N + col;
                if (EPI == 0) {
                    OutF[idx] = val;
                } else if (EPI == 1) {
                    OutF[idx] = Res[idx] + val;
                } else {
                    OutB[idx] = f2bf(0.5f * val * (1.0f + erff(val * 0.70710678118f)));
                }
            }
}

// ---------------------------------------------------------------------------
// RoPE + split qkv + write bf16 q / kv-cache.  One block per (b,t), 256 thr.
// qbf: [B,H,T,D]  kbf/vbf: [B,H,CAP,D]
// ---------------------------------------------------------------------------
__global__ __launch_bounds__(256) void rope_split(const float* __restrict__ qkv,
                                                  const int* __restrict__ offset,
                                                  const int* __restrict__ end_offset,
                                                  __bf16* __restrict__ qbf,
                                                  __bf16* __restrict__ kbf,
                                                  __bf16* __restrict__ vbf) {
    const int bt = blockIdx.x;
    const int b = bt / TT, t = bt % TT;
    const float pos = (float)(offset[b] + t);
    const int slot = (end_offset[b] + t) % CAP;
    const float* row = qkv + (size_t)bt * (3 * CC);
    const float lg = logf(10000.0f);

#pragma unroll
    for (int ip = 0; ip < (HH * DS) / 256; ++ip) {
        int p = threadIdx.x + 256 * ip;       // pair index over H*DS
        int h = p >> 5, jj = p & (DS - 1);
        float invf = expf(-((float)jj / DS) * lg);
        float ang = pos * invf;
        float cs = cosf(ang), sn = sinf(ang);

        int src = h * DD + 2 * jj;
        // q
        float qr = row[src], qi = row[src + 1];
        size_t qd = ((size_t)(b * HH + h) * TT + t) * DD + 2 * jj;
        qbf[qd]     = f2bf(qr * cs - qi * sn);
        qbf[qd + 1] = f2bf(qr * sn + qi * cs);
        // k
        float kr = row[CC + src], ki = row[CC + src + 1];
        size_t kd = ((size_t)(b * HH + h) * CAP + slot) * DD + 2 * jj;
        kbf[kd]     = f2bf(kr * cs - ki * sn);
        kbf[kd + 1] = f2bf(kr * sn + ki * cs);
    }
#pragma unroll
    for (int iv = 0; iv < CC / 256; ++iv) {
        int n = threadIdx.x + 256 * iv;       // h*D + d
        int h = n / DD, d = n % DD;
        vbf[((size_t)(b * HH + h) * CAP + slot) * DD + d] = f2bf(row[2 * CC + n]);
    }
}

// ---------------------------------------------------------------------------
// positions[b][s] per the reference cache-slot logic
// ---------------------------------------------------------------------------
__global__ __launch_bounds__(256) void make_positions(const int* __restrict__ end_offset,
                                                      int* __restrict__ positions) {
    int i = blockIdx.x * 256 + threadIdx.x;
    if (i >= BB * CAP) return;
    int b = i / CAP, s = i % CAP;
    int eo = end_offset[b];
    int last = eo + TT - 1;
    int end_index = last % CAP;
    int delta = s - end_index;
    int p = (delta <= 0) ? (last + delta) : (last + delta - CAP);
    if (s >= eo + TT) p = -1;
    positions[i] = p;
}

// ---------------------------------------------------------------------------
// Flash attention: block = 4 waves (128 thr), one block per (b,h,64 queries).
// Wave owns 16 query rows; streams key tiles of 32 with online softmax.
// QK^T and PV both via v_wmma_f32_16x16x32_bf16. Output -> obf [B,T,H*D] bf16.
// ---------------------------------------------------------------------------
__global__ __launch_bounds__(128) void attention(const __bf16* __restrict__ qbf,
                                                 const __bf16* __restrict__ kbf,
                                                 const __bf16* __restrict__ vbf,
                                                 const int* __restrict__ positions,
                                                 const int* __restrict__ offset,
                                                 const int* __restrict__ ctxp,
                                                 __bf16* __restrict__ obf) {
    __shared__ __bf16 Vt[64][40];       // transposed V tile: [d][k]
    __shared__ __bf16 Pl[4][16][40];    // per-wave P tile:   [row][k]

    const int nblk = TT / 64;
    const int bh = blockIdx.x / nblk;
    const int tb = blockIdx.x % nblk;
    const int b = bh / HH, h = bh % HH;
    const int wave = threadIdx.x >> 5;
    const int lane = threadIdx.x & 31;
    const int lr = lane & 15, lh = lane >> 4;
    const int t0 = tb * 64 + wave * 16;
    const int ctxv = ctxp[0];
    const int offb = offset[b];
    const float scale = 0.125f;   // 1/sqrt(64)

    // Q fragments (K=64 -> two 16x32 A-frags), straight from global
    FragU fq[2];
    const __bf16* qrow = qbf + ((size_t)(b * HH + h) * TT + (t0 + lr)) * DD;
#pragma unroll
    for (int ks = 0; ks < 2; ++ks) {
        fq[ks].u[0] = *(const uint4*)(qrow + ks * 32 + 8 * lh);
        fq[ks].u[1] = *(const uint4*)(qrow + ks * 32 + 8 * lh + 16);
    }

    float m[8], lsum[8];
#pragma unroll
    for (int r = 0; r < 8; ++r) { m[r] = -1e30f; lsum[r] = 0.f; }
    v8f o[4];
#pragma unroll
    for (int ni = 0; ni < 4; ++ni) o[ni] = vzero8();

    const __bf16* kbase = kbf + (size_t)(b * HH + h) * CAP * DD;
    const __bf16* vbase = vbf + (size_t)(b * HH + h) * CAP * DD;
    const int* posb = positions + b * CAP;

    for (int s0 = 0; s0 < CAP; s0 += 32) {
        // cooperative V transpose: 32x64 -> Vt[d][k]
#pragma unroll
        for (int i = 0; i < 16; ++i) {
            int idx = threadIdx.x + 128 * i;
            int k = idx & 31, d = idx >> 5;
            Vt[d][k] = vbase[(size_t)(s0 + k) * DD + d];
        }
        __syncthreads();

        // scores = q . k^T  (two 16-key subtiles x two K-steps)
        v8f sc[2] = {vzero8(), vzero8()};
#pragma unroll
        for (int sub = 0; sub < 2; ++sub) {
            const __bf16* krow = kbase + (size_t)(s0 + sub * 16 + lr) * DD;
#pragma unroll
            for (int ks = 0; ks < 2; ++ks) {
                FragU kb;
                kb.u[0] = *(const uint4*)(krow + ks * 32 + 8 * lh);
                kb.u[1] = *(const uint4*)(krow + ks * 32 + 8 * lh + 16);
                sc[sub] = wmma_bf16(fq[ks].v, kb.v, sc[sub]);
            }
        }

        const int p0 = posb[s0 + lr];
        const int p1 = posb[s0 + 16 + lr];

#pragma unroll
        for (int r = 0; r < 8; ++r) {
            int t = t0 + r + 8 * lh;
            int pq = offb + t;
            float v0 = sc[0][r] * scale, v1 = sc[1][r] * scale;
            int d0 = pq - p0, d1 = pq - p1;
            if (!(p0 >= 0 && d0 >= 0 && d0 < ctxv)) v0 = -3e38f;
            if (!(p1 >= 0 && d1 >= 0 && d1 < ctxv)) v1 = -3e38f;
            float mx = fmaxf(v0, v1);
#pragma unroll
            for (int off = 1; off < 16; off <<= 1)
                mx = fmaxf(mx, __shfl_xor(mx, off, 32));
            float mnew = fmaxf(fmaxf(m[r], mx), -1e30f);
            float f = __expf(m[r] - mnew);
            float e0 = __expf(v0 - mnew), e1 = __expf(v1 - mnew);
            float rs = e0 + e1;
#pragma unroll
            for (int off = 1; off < 16; off <<= 1)
                rs += __shfl_xor(rs, off, 32);
            lsum[r] = lsum[r] * f + rs;
            m[r] = mnew;
#pragma unroll
            for (int ni = 0; ni < 4; ++ni) o[ni][r] *= f;
            sc[0][r] = e0; sc[1][r] = e1;
            // stash P in LDS in A-matrix [row][k] layout
            Pl[wave][r + 8 * lh][lr]      = f2bf(e0);
            Pl[wave][r + 8 * lh][16 + lr] = f2bf(e1);
        }
        __syncthreads();

        // o += P (16x32) . V (32x64)
        FragU pa;
        pa.u[0] = *(const uint4*)&Pl[wave][lr][8 * lh];
        pa.u[1] = *(const uint4*)&Pl[wave][lr][8 * lh + 16];
#pragma unroll
        for (int ni = 0; ni < 4; ++ni) {
            FragU vb;
            vb.u[0] = *(const uint4*)&Vt[ni * 16 + lr][8 * lh];
            vb.u[1] = *(const uint4*)&Vt[ni * 16 + lr][8 * lh + 16];
            o[ni] = wmma_bf16(pa.v, vb.v, o[ni]);
        }
        __syncthreads();
    }

    // normalize + write [B,T,H*D] bf16
#pragma unroll
    for (int r = 0; r < 8; ++r) {
        float inv = (lsum[r] > 0.f) ? (1.0f / lsum[r]) : 0.f;
        int t = t0 + r + 8 * lh;
        __bf16* orow = obf + ((size_t)(b * TT + t) * HH + h) * DD;
#pragma unroll
        for (int ni = 0; ni < 4; ++ni)
            orow[ni * 16 + lr] = f2bf(o[ni][r] * inv);
    }
}

// ---------------------------------------------------------------------------
// Host-side orchestration
// ---------------------------------------------------------------------------
extern "C" void kernel_launch(void* const* d_in, const int* in_sizes, int n_in,
                              void* d_out, int out_size, void* d_ws, size_t ws_size,
                              hipStream_t stream) {
    (void)in_sizes; (void)n_in; (void)out_size; (void)ws_size;
    const float* x          = (const float*)d_in[0];
    const float* cache      = (const float*)d_in[1];
    const float* in_proj_w  = (const float*)d_in[2];
    const float* out_proj_w = (const float*)d_in[3];
    const float* ln1_w      = (const float*)d_in[4];
    const float* ln1_b      = (const float*)d_in[5];
    const float* ln2_w      = (const float*)d_in[6];
    const float* ln2_b      = (const float*)d_in[7];
    const float* lin1_w     = (const float*)d_in[8];
    const float* lin2_w     = (const float*)d_in[9];
    const int*   offset     = (const int*)d_in[10];
    const int*   end_offset = (const int*)d_in[11];
    const int*   ctx        = (const int*)d_in[12];
    float* out = (float*)d_out;

    const int M = BB * TT;                       // 2048
    const size_t nKV = (size_t)BB * HH * CAP * DD;

    char* w = (char*)d_ws;
    auto carve = [&](size_t bytes) { void* p = w; w += (bytes + 255) & ~(size_t)255; return p; };

    __bf16* wq  = (__bf16*)carve((size_t)3 * CC * CC * 2);
    __bf16* wo  = (__bf16*)carve((size_t)CC * CC * 2);
    __bf16* w1  = (__bf16*)carve((size_t)FF * CC * 2);
    __bf16* w2  = (__bf16*)carve((size_t)CC * FF * 2);
    __bf16* hb  = (__bf16*)carve((size_t)M * CC * 2);
    float*  qkv = (float*) carve((size_t)M * 3 * CC * 4);
    __bf16* qb  = (__bf16*)carve((size_t)BB * HH * TT * DD * 2);
    __bf16* kb  = (__bf16*)carve(nKV * 2);       // kb and vb contiguous
    __bf16* vb  = (__bf16*)carve(nKV * 2);
    __bf16* ob  = (__bf16*)carve((size_t)M * CC * 2);
    float*  x2  = (float*) carve((size_t)M * CC * 4);
    __bf16* h2  = (__bf16*)carve((size_t)M * CC * 2);
    __bf16* gb  = (__bf16*)carve((size_t)M * FF * 2);
    int*    pos = (int*)   carve((size_t)BB * CAP * 4);

    auto cvt = [&](const float* s, __bf16* d, size_t n) {
        int blocks = (int)((n + 255) / 256);
        cvt_f32_bf16<<<blocks, 256, 0, stream>>>(s, d, (int)n);
    };
    cvt(in_proj_w,  wq, (size_t)3 * CC * CC);
    cvt(out_proj_w, wo, (size_t)CC * CC);
    cvt(lin1_w,     w1, (size_t)FF * CC);
    cvt(lin2_w,     w2, (size_t)CC * FF);
    cvt(cache,      kb, 2 * nKV);                // keys then values (contiguous)

    make_positions<<<(BB * CAP + 255) / 256, 256, 0, stream>>>(end_offset, pos);

    // LN1 -> bf16
    layernorm_bf16<<<M, 256, 0, stream>>>(x, ln1_w, ln1_b, hb);

    // qkv = h @ in_proj_w^T   (2048 x 6144 x 2048), fp32 out
    gemm_bf16<0><<<dim3(3 * CC / 128, M / 128), 256, 0, stream>>>(
        hb, wq, qkv, nullptr, nullptr, M, 3 * CC, CC);

    // RoPE + split into q / kv-cache (bf16)
    rope_split<<<BB * TT, 256, 0, stream>>>(qkv, offset, end_offset, qb, kb, vb);

    // flash attention -> ob (B,T,C) bf16
    attention<<<BB * HH * (TT / 64), 128, 0, stream>>>(qb, kb, vb, pos, offset, ctx, ob);

    // x2 = x + attn @ out_proj_w^T
    gemm_bf16<1><<<dim3(CC / 128, M / 128), 256, 0, stream>>>(
        ob, wo, x2, nullptr, x, M, CC, CC);

    // LN2 -> bf16
    layernorm_bf16<<<M, 256, 0, stream>>>(x2, ln2_w, ln2_b, h2);

    // g = gelu(h2 @ lin1_w^T) -> bf16   (2048 x 8192 x 2048)
    gemm_bf16<2><<<dim3(FF / 128, M / 128), 256, 0, stream>>>(
        h2, w1, nullptr, gb, nullptr, M, FF, CC);

    // out = x2 + g @ lin2_w^T           (2048 x 2048 x 8192)
    gemm_bf16<1><<<dim3(CC / 128, M / 128), 256, 0, stream>>>(
        gb, w2, out, nullptr, x2, M, CC, FF);
}
